// SwitchHeadAttention_16939351016118
// MI455X (gfx1250) — matrix operand order repaired and checked by reading, not compiled, not run on hardware
//
#include <hip/hip_runtime.h>
#include <hip/hip_bf16.h>

// SwitchHead attention for MI455X (gfx1250, wave32, WMMA).
// B=4, T=2048, DIM=1024, H=8, DH=64, TOPK=3.

typedef __attribute__((ext_vector_type(16))) __bf16 v16bf;
typedef __attribute__((ext_vector_type(8)))  __bf16 v8bf;
typedef __attribute__((ext_vector_type(8)))  float  v8f;
typedef int v4i __attribute__((vector_size(16)));
typedef __attribute__((address_space(1))) v4i gv4i;   // global b128 payload
typedef __attribute__((address_space(3))) v4i lv4i;   // LDS b128 payload

#define WMMA_BF16(a, b, c) \
  __builtin_amdgcn_wmma_f32_16x16x32_bf16(false, (a), false, (b), (short)0, (c), false, false)

#if __has_builtin(__builtin_amdgcn_global_load_async_to_lds_b128) && \
    __has_builtin(__builtin_amdgcn_s_wait_asynccnt)
#define HAVE_ASYNC_LDS 1
#endif

static __device__ __forceinline__ v8bf ld8(const __bf16* p) { return *(const v8bf*)p; }

static __device__ __forceinline__ v16bf mk16(v8bf lo, v8bf hi) {
  v16bf r;
#pragma unroll
  for (int i = 0; i < 8; ++i) { r[i] = lo[i]; r[i + 8] = hi[i]; }
  return r;
}

#define NB 4
#define NT 2048
#define NDIM 1024
#define NH 8
#define NDH 64
#define NM (NB * NT)        // 8192 tokens
#define NQKV 1536           // 512 (Q) + 512 (K) + 512 (V) output columns

// ---------------------------------------------------------------- prep kernels
__global__ void k_cvt_x(const float* __restrict__ x, __bf16* __restrict__ xbf) {
  int row = blockIdx.x;
  int t = threadIdx.x;                       // 256 threads * 4 floats = 1024
  const float4 v = ((const float4*)(x + (size_t)row * NDIM))[t];
  __bf16* d = xbf + (size_t)row * NDIM + t * 4;
  d[0] = (__bf16)v.x; d[1] = (__bf16)v.y; d[2] = (__bf16)v.z; d[3] = (__bf16)v.w;
}

// Wt[n][k]: n<512 -> Wq[:,n] * DH^-0.5 (q scale folded), else Wkv[:,n-512]
__global__ void k_prep_w(const float* __restrict__ Wq, const float* __restrict__ Wkv,
                         __bf16* __restrict__ Wt) {
  int n = blockIdx.x;                        // 0..1535
  for (int k = threadIdx.x; k < NDIM; k += blockDim.x) {
    float v = (n < 512) ? Wq[(size_t)k * 512 + n] * 0.125f
                        : Wkv[(size_t)k * 1024 + (n - 512)];
    Wt[(size_t)n * NDIM + k] = (__bf16)v;
  }
}

// WoT[h][n][k] = Wo[h][k][n]  (bf16, K-contiguous for B fragments)
__global__ void k_prep_wo(const float* __restrict__ Wo, __bf16* __restrict__ WoT) {
  int hn = blockIdx.x;                       // h*1024 + n
  int h = hn >> 10, n = hn & 1023;
  int k = threadIdx.x;                       // 64
  WoT[(size_t)hn * NDH + k] = (__bf16)Wo[((size_t)(h * NDH + k)) * NDIM + n];
}

// ------------------------------------------------------------- gating (top-3)
__global__ void k_scores(const float* __restrict__ x, const float* __restrict__ Ws,
                         float* __restrict__ scores) {
  int w = threadIdx.x >> 5, lane = threadIdx.x & 31;
  int tok = blockIdx.x * 8 + w;
  float a[NH];
#pragma unroll
  for (int h = 0; h < NH; ++h) a[h] = 0.f;
  const float* xr = x + (size_t)tok * NDIM;
  for (int k = lane; k < NDIM; k += 32) {
    float xv = xr[k];
    const float* wr = Ws + (size_t)k * NH;
#pragma unroll
    for (int h = 0; h < NH; ++h) a[h] += xv * wr[h];
  }
#pragma unroll
  for (int h = 0; h < NH; ++h)
#pragma unroll
    for (int m = 16; m >= 1; m >>= 1) a[h] += __shfl_xor(a[h], m, 32);
  if (lane == 0) {
    float s[NH], o[NH];
#pragma unroll
    for (int h = 0; h < NH; ++h) { s[h] = 1.f / (1.f + __expf(-a[h])); o[h] = 0.f; }
    for (int it = 0; it < 3; ++it) {
      int best = 0; float bv = s[0];
#pragma unroll
      for (int h = 1; h < NH; ++h) if (s[h] > bv) { bv = s[h]; best = h; }
      o[best] = bv; s[best] = -1e30f;
    }
    float* d = scores + (size_t)tok * NH;
#pragma unroll
    for (int h = 0; h < NH; ++h) d[h] = o[h];
  }
}

// ------------------------------------------------- fused QKV projection (WMMA)
// Ping-pong register double-buffer (unroll by 2): fragment sets 0/1 alternate
// as load target vs WMMA source, so no per-iteration register copies.
__global__ void __launch_bounds__(256)
k_gemm_qkv(const __bf16* __restrict__ xbf, const __bf16* __restrict__ Wt,
           __bf16* __restrict__ Qbf, __bf16* __restrict__ Kbf, __bf16* __restrict__ Vbf) {
  int w = threadIdx.x >> 5, lane = threadIdx.x & 31;
  int wm = w & 3, wn = w >> 2;
  int row0 = blockIdx.x * 64 + wm * 16;
  int col0 = blockIdx.y * 128 + wn * 64;
  int half = lane >> 4, ln = lane & 15;
  int half8 = half * 8, half16 = half * 16;
  v8f acc[4] = {};

  const __bf16* abase = xbf + (size_t)(row0 + ln) * NDIM;
  const __bf16* bbase[4];
#pragma unroll
  for (int f = 0; f < 4; ++f)
    bbase[f] = Wt + (size_t)(col0 + f * 16 + ln) * NDIM + half16;

  v16bf a0, a1, b0[4], b1[4];
  a0 = mk16(ld8(abase + half16), ld8(abase + 8 + half16));
#pragma unroll
  for (int f = 0; f < 4; ++f) b0[f] = mk16(ld8(bbase[f]), ld8(bbase[f] + 8));

  int k = 0;
  for (; k + 64 < NDIM; k += 64) {
    __builtin_prefetch(abase + k + 128, 0, 0);    // global_prefetch_b8
    a1 = mk16(ld8(abase + k + 32 + half16), ld8(abase + k + 40 + half16));
#pragma unroll
    for (int f = 0; f < 4; ++f)
      b1[f] = mk16(ld8(bbase[f] + k + 32), ld8(bbase[f] + k + 40));
#pragma unroll
    for (int f = 0; f < 4; ++f) acc[f] = WMMA_BF16(a0, b0[f], acc[f]);

    a0 = mk16(ld8(abase + k + 64 + half16), ld8(abase + k + 72 + half16));
#pragma unroll
    for (int f = 0; f < 4; ++f)
      b0[f] = mk16(ld8(bbase[f] + k + 64), ld8(bbase[f] + k + 72));
#pragma unroll
    for (int f = 0; f < 4; ++f) acc[f] = WMMA_BF16(a1, b1[f], acc[f]);
  }
  // k == NDIM-64: set0 holds k, stage k+32 then drain both.
  a1 = mk16(ld8(abase + k + 32 + half16), ld8(abase + k + 40 + half16));
#pragma unroll
  for (int f = 0; f < 4; ++f)
    b1[f] = mk16(ld8(bbase[f] + k + 32), ld8(bbase[f] + k + 40));
#pragma unroll
  for (int f = 0; f < 4; ++f) acc[f] = WMMA_BF16(a0, b0[f], acc[f]);
#pragma unroll
  for (int f = 0; f < 4; ++f) acc[f] = WMMA_BF16(a1, b1[f], acc[f]);

#pragma unroll
  for (int f = 0; f < 4; ++f) {
    int n = col0 + f * 16 + ln;
#pragma unroll
    for (int r = 0; r < 8; ++r) {
      int tok = row0 + r + half8;
      int b = tok >> 11, t = tok & (NT - 1);
      __bf16 v = (__bf16)acc[f][r];
      if (n < 512) {
        int h = n >> 6, d = n & 63;
        Qbf[(((size_t)(b * NH + h)) * NT + t) * NDH + d] = v;
      } else if (n < 1024) {
        int nn = n - 512, h = nn >> 6, d = nn & 63;
        Kbf[(((size_t)(b * NH + h)) * NT + t) * NDH + d] = v;
      } else {
        int nn = n - 1024, h = nn >> 6, d = nn & 63;
        Vbf[(((size_t)(b * NH + h)) * NT + t) * NDH + d] = v;
      }
    }
  }
}

// --------------------------------------------------- flash attention (WMMA+LDS)
// Double-buffered tiles: K staged memory->LDS with GLOBAL_LOAD_ASYNC_TO_LDS_B128
// for tile j+1 while tile j's WMMAs/softmax run; V transposes via VGPRs.
__global__ void __launch_bounds__(256)
k_attn(const __bf16* __restrict__ Qbf, const __bf16* __restrict__ Kbf,
       const __bf16* __restrict__ Vbf, __bf16* __restrict__ Obf) {
  __shared__ __bf16 lK[2][32 * 64];    // key tiles, d-contiguous
  __shared__ __bf16 lVT[2][64 * 32];   // value tiles, transposed (d-major)
  __shared__ __bf16 lP[8][16 * 32];    // per-wave softmax P staging
  int w = threadIdx.x >> 5, lane = threadIdx.x & 31, tid = threadIdx.x;
  int bh = blockIdx.y;
  int qrow0 = blockIdx.x * 128 + w * 16;
  int half = lane >> 4, ln = lane & 15;
  int jj = tid >> 3, c = tid & 7;
  const size_t kvrow = ((size_t)bh * NT + jj) * NDH + c * 8;

  const __bf16* qb = Qbf + (((size_t)bh * NT) + qrow0 + ln) * NDH;
  v16bf aQ0 = mk16(ld8(qb + half * 16),      ld8(qb + 8 + half * 16));
  v16bf aQ1 = mk16(ld8(qb + 32 + half * 16), ld8(qb + 40 + half * 16));

  v8f o[4] = {};
  float m_[8], l_[8];
#pragma unroll
  for (int r = 0; r < 8; ++r) { m_[r] = -1e30f; l_[r] = 0.f; }

  // prologue: stage tile 0 into buffer 0
  {
    const size_t kvbase = kvrow;               // j0 = 0
#ifdef HAVE_ASYNC_LDS
    __builtin_amdgcn_global_load_async_to_lds_b128(
        (gv4i*)(Kbf + kvbase), (lv4i*)&lK[0][jj * 64 + c * 8], 0, 0);
#else
    *(v8bf*)&lK[0][jj * 64 + c * 8] = ld8(Kbf + kvbase);
#endif
    v8bf vv = ld8(Vbf + kvbase);
#pragma unroll
    for (int i = 0; i < 8; ++i) lVT[0][(c * 8 + i) * 32 + jj] = vv[i];
#ifdef HAVE_ASYNC_LDS
    __builtin_amdgcn_s_wait_asynccnt(0);
#endif
    __syncthreads();
  }

  int cur = 0;
  for (int j0 = 0; j0 < NT; j0 += 32, cur ^= 1) {
    // stage next tile into the other buffer; overlaps with this tile's math
    if (j0 + 32 < NT) {
      int nb = cur ^ 1;
      const size_t kvbase = kvrow + (size_t)(j0 + 32) * NDH;
#ifdef HAVE_ASYNC_LDS
      __builtin_amdgcn_global_load_async_to_lds_b128(
          (gv4i*)(Kbf + kvbase), (lv4i*)&lK[nb][jj * 64 + c * 8], 0, 0);
#else
      *(v8bf*)&lK[nb][jj * 64 + c * 8] = ld8(Kbf + kvbase);
#endif
      v8bf vv = ld8(Vbf + kvbase);
#pragma unroll
      for (int i = 0; i < 8; ++i) lVT[nb][(c * 8 + i) * 32 + jj] = vv[i];
    }

    // S = Q @ K^T for 32-key tile (two 16x16 C frags, K-dim 64 = 2 steps)
    v8f s0 = {}, s1 = {};
    {
      const __bf16* p; v16bf b;
      p = &lK[cur][ln * 64 + half * 16];             b = mk16(ld8(p), ld8(p + 8)); s0 = WMMA_BF16(aQ0, b, s0);
      p = &lK[cur][ln * 64 + 32 + half * 16];        b = mk16(ld8(p), ld8(p + 8)); s0 = WMMA_BF16(aQ1, b, s0);
      p = &lK[cur][(16 + ln) * 64 + half * 16];      b = mk16(ld8(p), ld8(p + 8)); s1 = WMMA_BF16(aQ0, b, s1);
      p = &lK[cur][(16 + ln) * 64 + 32 + half * 16]; b = mk16(ld8(p), ld8(p + 8)); s1 = WMMA_BF16(aQ1, b, s1);
    }

    // online softmax; row r lives in lanes [0..15] (M=r) / [16..31] (M=r+8)
#pragma unroll
    for (int r = 0; r < 8; ++r) {
      float v = fmaxf(s0[r], s1[r]);
#pragma unroll
      for (int msk = 8; msk >= 1; msk >>= 1) v = fmaxf(v, __shfl_xor(v, msk, 16));
      float mn = fmaxf(m_[r], v);
      float scl = __expf(m_[r] - mn);
      float p0 = __expf(s0[r] - mn);
      float p1 = __expf(s1[r] - mn);
      float rs = p0 + p1;
#pragma unroll
      for (int msk = 8; msk >= 1; msk >>= 1) rs += __shfl_xor(rs, msk, 16);
      l_[r] = l_[r] * scl + rs;
      m_[r] = mn;
#pragma unroll
      for (int f = 0; f < 4; ++f) o[f][r] *= scl;
      lP[w][(r + half * 8) * 32 + ln]      = (__bf16)p0;
      lP[w][(r + half * 8) * 32 + 16 + ln] = (__bf16)p1;
    }
    __syncthreads();

    // O += P @ V  (A frag of P reloaded from LDS in A layout)
    const __bf16* pp = &lP[w][ln * 32 + half * 16];
    v16bf aP = mk16(ld8(pp), ld8(pp + 8));
#pragma unroll
    for (int f = 0; f < 4; ++f) {
      const __bf16* vp = &lVT[cur][(f * 16 + ln) * 32 + half * 16];
      v16bf bV = mk16(ld8(vp), ld8(vp + 8));
      o[f] = WMMA_BF16(aP, bV, o[f]);
    }
#ifdef HAVE_ASYNC_LDS
    __builtin_amdgcn_s_wait_asynccnt(0);
#endif
    __syncthreads();   // next tile resident; all reads of cur retired
  }

#pragma unroll
  for (int f = 0; f < 4; ++f) {
    int d = f * 16 + ln;
#pragma unroll
    for (int r = 0; r < 8; ++r) {
      int q = qrow0 + r + half * 8;
      Obf[(((size_t)bh * NT) + q) * NDH + d] = (__bf16)(o[f][r] / l_[r]);
    }
  }
}

// ------------------------------- gated per-head output experts + bias (WMMA)
__global__ void __launch_bounds__(256)
k_outproj(const __bf16* __restrict__ Obf, const __bf16* __restrict__ WoT,
          const float* __restrict__ scores, const float* __restrict__ bo,
          float* __restrict__ out) {
  int w = threadIdx.x >> 5, lane = threadIdx.x & 31;
  int wm = w & 3, wn = w >> 2;
  int row0 = blockIdx.x * 64 + wm * 16;
  int col0 = blockIdx.y * 128 + wn * 64;
  int half = lane >> 4, ln = lane & 15;
  v8f acc[4] = {};
  int atok = row0 + ln;
  int ab = atok >> 11, at = atok & (NT - 1);
#pragma unroll
  for (int h = 0; h < NH; ++h) {
    float sc = scores[(size_t)atok * NH + h];        // gate folded into A rows
    const __bf16* ob = Obf + (((size_t)(ab * NH + h)) * NT + at) * NDH;
#pragma unroll
    for (int kk = 0; kk < NDH; kk += 32) {
      v8bf lo = ld8(ob + kk + half * 16);
      v8bf hi = ld8(ob + kk + 8 + half * 16);
      v16bf a;
#pragma unroll
      for (int i = 0; i < 8; ++i) {
        a[i]     = (__bf16)((float)lo[i] * sc);
        a[i + 8] = (__bf16)((float)hi[i] * sc);
      }
#pragma unroll
      for (int f = 0; f < 4; ++f) {
        const __bf16* wp = WoT + ((size_t)h * NDIM + col0 + f * 16 + ln) * NDH + kk + half * 16;
        v16bf b = mk16(ld8(wp), ld8(wp + 8));
        acc[f] = WMMA_BF16(a, b, acc[f]);
      }
    }
  }
#pragma unroll
  for (int f = 0; f < 4; ++f) {
    int n = col0 + f * 16 + ln;
    float bo8[NH];
#pragma unroll
    for (int h = 0; h < NH; ++h) bo8[h] = bo[(size_t)h * NDIM + n];
#pragma unroll
    for (int r = 0; r < 8; ++r) {
      int tok = row0 + r + half * 8;
      float v = acc[f][r];
      const float* sp = scores + (size_t)tok * NH;
#pragma unroll
      for (int h = 0; h < NH; ++h) v += sp[h] * bo8[h];
      out[(size_t)tok * NDIM + n] = v;
    }
  }
}

// ----------------------------------------------------------------- launcher
extern "C" void kernel_launch(void* const* d_in, const int* in_sizes, int n_in,
                              void* d_out, int out_size, void* d_ws, size_t ws_size,
                              hipStream_t stream) {
  (void)in_sizes; (void)n_in; (void)out_size; (void)ws_size;
  const float* x   = (const float*)d_in[0];
  const float* Wq  = (const float*)d_in[1];
  const float* Wkv = (const float*)d_in[2];
  const float* Ws  = (const float*)d_in[3];
  const float* Wo  = (const float*)d_in[4];
  const float* bo  = (const float*)d_in[5];
  float* out = (float*)d_out;

  char* p = (char*)d_ws;
  __bf16* xbf = (__bf16*)p; p += (size_t)NM * NDIM * 2;           // 16 MB
  __bf16* Wt  = (__bf16*)p; p += (size_t)NQKV * NDIM * 2;         //  3 MB
  __bf16* WoT = (__bf16*)p; p += (size_t)NH * NDIM * NDH * 2;     //  1 MB
  __bf16* Qbf = (__bf16*)p; p += (size_t)NM * NDH * 2;            //  1 MB
  __bf16* Kbf = (__bf16*)p; p += (size_t)NM * NDH * 2;            //  1 MB
  __bf16* Vbf = (__bf16*)p; p += (size_t)NM * NDH * 2;            //  1 MB
  __bf16* Obf = (__bf16*)p; p += (size_t)NM * NDH * 2;            //  1 MB
  float* scores = (float*)p; p += (size_t)NM * NH * 4;            // .25 MB

  k_cvt_x  <<<NM, 256, 0, stream>>>(x, xbf);
  k_prep_w <<<NQKV, 256, 0, stream>>>(Wq, Wkv, Wt);
  k_prep_wo<<<NH * NDIM, 64, 0, stream>>>(Wo, WoT);
  k_scores <<<NM / 8, 256, 0, stream>>>(x, Ws, scores);
  k_gemm_qkv<<<dim3(NM / 64, NQKV / 128), 256, 0, stream>>>(xbf, Wt, Qbf, Kbf, Vbf);
  k_attn   <<<dim3(NT / 128, NB * NH), 256, 0, stream>>>(Qbf, Kbf, Vbf, Obf);
  k_outproj<<<dim3(NM / 64, NDIM / 128), 256, 0, stream>>>(Obf, WoT, scores, bo, out);
}
